// PrecursorPredictionGNN_18983755448848
// MI455X (gfx1250) — matrix operand order repaired and checked by reading, not compiled, not run on hardware
//
#include <hip/hip_runtime.h>
#include <hip/hip_bf16.h>
#include <math.h>

// ---------------------------------------------------------------------------
// GATv2 GNN inference for MI455X (gfx1250).
// Dense node GEMMs -> bf16 WMMA (v_wmma_f32_16x16x32_bf16, f32 accumulate),
// async global->LDS staging (ASYNCcnt) when the toolchain exposes it,
// b128 LDS fragment loads. Sparse edge softmax/aggregation -> VALU + f32
// global atomics (L2-resident).
// ---------------------------------------------------------------------------

#define HEADS 4
#define HID 128
#define LAYERS 3
#define NEG_SLOPE 0.2f

#if __has_builtin(__builtin_amdgcn_global_load_async_to_lds_b128)
#define HAVE_ASYNC_LDS 1
#endif

typedef __attribute__((ext_vector_type(16))) __bf16 v16bf;
typedef __attribute__((ext_vector_type(8)))  float  v8f;
typedef __attribute__((ext_vector_type(4)))  int    v4i;

#define AS1 __attribute__((address_space(1)))
#define AS3 __attribute__((address_space(3)))

union Frag16 {
    v16bf v;
    unsigned int u[8];
};

__device__ __forceinline__ unsigned short f32_to_bf16_rne(float f) {
    unsigned int u = __float_as_uint(f);
    unsigned int r = u + 0x7FFFu + ((u >> 16) & 1u);  // round-to-nearest-even
    return (unsigned short)(r >> 16);
}

// sign-aware float atomic max via integer punning (works for mixed signs)
__device__ __forceinline__ void atomicMaxF32(float* addr, float val) {
    if (val >= 0.0f) {
        atomicMax((int*)addr, __float_as_int(val));
    } else {
        atomicMin((unsigned int*)addr, __float_as_uint(val));
    }
}

// f32 -> bf16 bulk conversion, 4 elements/thread, vectorized
__global__ __launch_bounds__(256)
void cvt_f32_to_bf16_kernel(const float* __restrict__ s,
                            unsigned short* __restrict__ d, long long n) {
    long long i = ((long long)blockIdx.x * 256 + threadIdx.x) * 4;
    if (i + 3 < n) {
        float4 f = *(const float4*)(s + i);
        unsigned int lo = (unsigned int)f32_to_bf16_rne(f.x) |
                          ((unsigned int)f32_to_bf16_rne(f.y) << 16);
        unsigned int hi = (unsigned int)f32_to_bf16_rne(f.z) |
                          ((unsigned int)f32_to_bf16_rne(f.w) << 16);
        uint2 q; q.x = lo; q.y = hi;
        *(uint2*)(d + i) = q;
    } else {
        for (; i < n; ++i) d[i] = f32_to_bf16_rne(s[i]);
    }
}

// ---------------------------------------------------------------------------
// Tiled GEMM: C[M x Nout] = act(A_bf16[M x K] * W_bf16[K x Nout] + bias)
// Block = 256 threads = 8 waves. Block tile: 128(M) x 64(N). K chunk = 32.
// Each wave: one 16-row strip, four 16x16 WMMA accumulators.
// Staging: A tile 2x async b128 global->LDS per thread (branch-free clamp);
//          B tile 1x global b128 per thread, stored transposed [n][k].
// Fragments: 2x ds_load_b128 each (row stride 40 ushorts = 80 B: 16B-aligned,
//            20-bank stride -> rows 0..15 on distinct banks).
// Optionally writes a bf16 mirror of C for the next GEMM in the chain.
// ---------------------------------------------------------------------------
#define MT 128
#define NT 64
#define KC 32
#define AS 40
#define BS 40

__global__ __launch_bounds__(256)
void gemm_bf16_wmma(const unsigned short* __restrict__ A,
                    const unsigned short* __restrict__ W,
                    const float* __restrict__ bias,
                    float* __restrict__ C, unsigned short* __restrict__ Cbf,
                    int M, int K, int Nout, int do_relu) {
    __shared__ unsigned short Alds[MT * AS];   // 10.0 KB
    __shared__ unsigned short Blds[NT * BS];   //  5.0 KB

    const int tid  = threadIdx.x;
    const int wave = tid >> 5;
    const int lane = tid & 31;
    const int m0   = blockIdx.x * MT;
    const int n0   = blockIdx.y * NT;

    const int mfrag = lane & 15;   // M (A frag) or N (B frag) index within tile
    const int kh    = lane >> 4;   // K-half selector

    v8f acc[4];
#pragma unroll
    for (int nt = 0; nt < 4; ++nt)
#pragma unroll
        for (int i = 0; i < 8; ++i) acc[nt][i] = 0.0f;

    for (int k0 = 0; k0 < K; k0 += KC) {
        __syncthreads();
        // stage A tile (128x32 bf16): 512 x 16B slots, 2 per thread, no branches
#pragma unroll
        for (int i = 0; i < 2; ++i) {
            int slot = i * 256 + tid;
            int r = slot >> 2, c8 = slot & 3;
            int gr = m0 + r;
            gr = (gr < M) ? gr : (M - 1);   // clamp: padding rows never stored
            const unsigned short* gsrc = A + (size_t)gr * K + k0 + c8 * 8;
            unsigned short* ldst = &Alds[r * AS + c8 * 8];
#ifdef HAVE_ASYNC_LDS
            __builtin_amdgcn_global_load_async_to_lds_b128(
                (AS1 v4i*)gsrc, (AS3 v4i*)ldst, 0, 0);
#else
            *(uint4*)ldst = *(const uint4*)gsrc;
#endif
        }
        // stage B tile (32x64 bf16), transposed into [n][k]
        {
            int k = tid >> 3, n8 = tid & 7;
            uint4 q = *(const uint4*)(W + (size_t)(k0 + k) * Nout + n0 + n8 * 8);
            unsigned int vv[4] = {q.x, q.y, q.z, q.w};
#pragma unroll
            for (int j = 0; j < 8; ++j) {
                unsigned short us = (unsigned short)(vv[j >> 1] >> ((j & 1) * 16));
                Blds[(n8 * 8 + j) * BS + k] = us;
            }
        }
#ifdef HAVE_ASYNC_LDS
#if __has_builtin(__builtin_amdgcn_s_wait_asynccnt)
        __builtin_amdgcn_s_wait_asynccnt(0);
#else
        asm volatile("s_wait_asynccnt 0x0" ::: "memory");
#endif
#endif
        __syncthreads();

        // A fragment: lane(m = mfrag, khalf = kh); u[0..3]=K kh*8..+7,
        // u[4..7]=K 16+kh*8..+7 -> two contiguous 16B runs
        Frag16 a;
        {
            const int base = (wave * 16 + mfrag) * AS + kh * 8;
            *(uint4*)&a.u[0] = *(const uint4*)&Alds[base];
            *(uint4*)&a.u[4] = *(const uint4*)&Alds[base + 16];
        }
        // hoist all four B fragments so dscnt waits stagger across the WMMAs
        Frag16 bfr[4];
#pragma unroll
        for (int nt = 0; nt < 4; ++nt) {
            const int bb = (nt * 16 + mfrag) * BS + kh * 8;
            *(uint4*)&bfr[nt].u[0] = *(const uint4*)&Blds[bb];
            *(uint4*)&bfr[nt].u[4] = *(const uint4*)&Blds[bb + 16];
        }
#pragma unroll
        for (int nt = 0; nt < 4; ++nt) {
            acc[nt] = __builtin_amdgcn_wmma_f32_16x16x32_bf16(
                false, a.v, false, bfr[nt].v, (short)0, acc[nt], false, false);
        }
    }

    // epilogue: C/D layout -> lane n = mfrag, VGPR v -> row (v + 8*kh)
#pragma unroll
    for (int nt = 0; nt < 4; ++nt) {
        int n = n0 + nt * 16 + mfrag;
        float bv = bias[n];
#pragma unroll
        for (int vv = 0; vv < 8; ++vv) {
            int m = m0 + wave * 16 + vv + 8 * kh;
            if (m < M) {
                float val = acc[nt][vv] + bv;
                if (do_relu) val = fmaxf(val, 0.0f);
                C[(size_t)m * Nout + n] = val;
                if (Cbf) Cbf[(size_t)m * Nout + n] = f32_to_bf16_rne(val);
            }
        }
    }
}

// ---------------------------------------------------------------------------
// Edge kernels (wave32 per edge)
// ---------------------------------------------------------------------------

// e[edge][h] = sum_c att[h][c] * leaky_relu(xl[src][h][c] + xr[dst][h][c])
// also segment-max into emax[dst][h]
__global__ __launch_bounds__(256)
void edge_scores_kernel(const int* __restrict__ ei,
                        const float* __restrict__ xl, const float* __restrict__ xr,
                        const float* __restrict__ att,
                        float* __restrict__ e_out, float* __restrict__ emax, int E) {
    int edge = (blockIdx.x * 256 + threadIdx.x) >> 5;
    int lane = threadIdx.x & 31;
    if (edge >= E) return;
    int s = ei[edge];
    int d = ei[E + edge];
    const float* pl = xl + (size_t)s * (HEADS * HID);
    const float* pr = xr + (size_t)d * (HEADS * HID);
#pragma unroll
    for (int h = 0; h < HEADS; ++h) {
        float sum = 0.0f;
#pragma unroll
        for (int j = 0; j < 4; ++j) {
            int c = lane + j * 32;
            float m = pl[h * HID + c] + pr[h * HID + c];
            m = (m > 0.0f) ? m : NEG_SLOPE * m;
            sum += att[h * HID + c] * m;
        }
#pragma unroll
        for (int off = 16; off > 0; off >>= 1) sum += __shfl_xor(sum, off, 32);
        if (lane == 0) {
            e_out[(size_t)edge * HEADS + h] = sum;
            atomicMaxF32(&emax[(size_t)d * HEADS + h], sum);
        }
    }
}

// ex = exp(e - emax[dst]); denom[dst] += ex; (e buffer rewritten in place)
__global__ __launch_bounds__(256)
void edge_expsum_kernel(const int* __restrict__ ei, float* __restrict__ e_buf,
                        const float* __restrict__ emax, float* __restrict__ denom,
                        int E) {
    int i = blockIdx.x * 256 + threadIdx.x;
    if (i >= E * HEADS) return;
    int edge = i >> 2;           // HEADS == 4
    int h = i & 3;
    int d = ei[E + edge];
    float ex = __expf(e_buf[i] - emax[(size_t)d * HEADS + h]);
    e_buf[i] = ex;
    atomicAdd(&denom[(size_t)d * HEADS + h], ex);
}

// acc[dst][c] += 0.25 * sum_h alpha[h] * xl[src][h][c]
__global__ __launch_bounds__(256)
void edge_aggregate_kernel(const int* __restrict__ ei, const float* __restrict__ xl,
                           const float* __restrict__ e_buf,
                           const float* __restrict__ denom,
                           float* __restrict__ acc, int E) {
    int edge = (blockIdx.x * 256 + threadIdx.x) >> 5;
    int lane = threadIdx.x & 31;
    if (edge >= E) return;
    int s = ei[edge];
    int d = ei[E + edge];
    float alpha[HEADS];
#pragma unroll
    for (int h = 0; h < HEADS; ++h) {
        float ex = e_buf[(size_t)edge * HEADS + h];
        float dn = denom[(size_t)d * HEADS + h];
        alpha[h] = ex / fmaxf(dn, 1e-16f);
    }
    const float* pl = xl + (size_t)s * (HEADS * HID);
#pragma unroll
    for (int j = 0; j < 4; ++j) {
        int c = lane + j * 32;
        float v = 0.0f;
#pragma unroll
        for (int h = 0; h < HEADS; ++h) v += alpha[h] * pl[h * HID + c];
        atomicAdd(&acc[(size_t)d * HID + c], 0.25f * v);
    }
}

// h = relu(h + acc + conv_bias); also refresh the bf16 mirror of h
__global__ __launch_bounds__(256)
void node_update_kernel(float* __restrict__ h, unsigned short* __restrict__ h_bf,
                        const float* __restrict__ acc,
                        const float* __restrict__ cbias, int N) {
    int i = blockIdx.x * 256 + threadIdx.x;
    if (i >= N * HID) return;
    float v = h[i] + acc[i] + cbias[i & (HID - 1)];
    v = fmaxf(v, 0.0f);
    h[i] = v;
    h_bf[i] = f32_to_bf16_rne(v);
}

__global__ __launch_bounds__(256)
void fill_f32_kernel(float* __restrict__ p, long long n, float v) {
    long long i = (long long)blockIdx.x * 256 + threadIdx.x;
    if (i < n) p[i] = v;
}

// scores = clip(z2 @ p3_W + p3_b, -15, 15)   (64 -> 1 GEMV, thread per node)
__global__ __launch_bounds__(256)
void p3_clip_kernel(const float* __restrict__ z2, const float* __restrict__ w,
                    const float* __restrict__ b, float* __restrict__ out, int N) {
    int i = blockIdx.x * 256 + threadIdx.x;
    if (i >= N) return;
    const float* z = z2 + (size_t)i * 64;
    float s = b[0];
#pragma unroll
    for (int c = 0; c < 64; ++c) s += z[c] * w[c];
    out[i] = fminf(fmaxf(s, -15.0f), 15.0f);
}

// ---------------------------------------------------------------------------

extern "C" void kernel_launch(void* const* d_in, const int* in_sizes, int n_in,
                              void* d_out, int out_size, void* d_ws, size_t ws_size,
                              hipStream_t stream) {
    const float* x      = (const float*)d_in[0];
    const int*   ei     = (const int*)d_in[1];
    // d_in[2] = batch (unused, single graph)
    const float* emb_W  = (const float*)d_in[3];
    const float* emb_b  = (const float*)d_in[4];
    const float* Wl     = (const float*)d_in[5];
    const float* bl     = (const float*)d_in[6];
    const float* Wr     = (const float*)d_in[7];
    const float* br     = (const float*)d_in[8];
    const float* att    = (const float*)d_in[9];
    const float* cbias  = (const float*)d_in[10];
    const float* p1_W   = (const float*)d_in[11];
    const float* p1_b   = (const float*)d_in[12];
    const float* p2_W   = (const float*)d_in[13];
    const float* p2_b   = (const float*)d_in[14];
    const float* p3_W   = (const float*)d_in[15];
    const float* p3_b   = (const float*)d_in[16];

    const int N = in_sizes[0] / 64;     // F_IN = 64
    const int E = in_sizes[1] / 2;
    const int F_IN = 64;

    // workspace layout: f32 region, then bf16 (ushort) region
    float* ws = (float*)d_ws;
    size_t o = 0;
    float* h    = ws + o; o += (size_t)N * HID;
    float* xl   = ws + o; o += (size_t)N * HEADS * HID;
    float* xr   = ws + o; o += (size_t)N * HEADS * HID;
    float* ebuf = ws + o; o += (size_t)E * HEADS;      // scores -> exp; later z2
    float* emax = ws + o; o += (size_t)N * HEADS;
    float* den  = ws + o; o += (size_t)N * HEADS;
    float* acc  = ws + o; o += (size_t)N * HID;        // aggregation; later z1

    unsigned short* us = (unsigned short*)(ws + o);
    size_t u = 0;
    unsigned short* x_bf   = us + u; u += (size_t)N * F_IN;
    unsigned short* h_bf   = us + u; u += (size_t)N * HID;
    unsigned short* z1_bf  = us + u; u += (size_t)N * HID;
    unsigned short* wb_emb = us + u; u += (size_t)F_IN * HID;
    unsigned short* wb_Wl  = us + u; u += (size_t)LAYERS * HID * HEADS * HID;
    unsigned short* wb_Wr  = us + u; u += (size_t)LAYERS * HID * HEADS * HID;
    unsigned short* wb_p1  = us + u; u += (size_t)HID * HID;
    unsigned short* wb_p2  = us + u; u += (size_t)HID * (HID / 2);
    (void)ws_size; (void)n_in; (void)out_size;

    const dim3 blk(256);
    const int mblocks = (N + MT - 1) / MT;

    // bf16 conversions (every call: deterministic, graph-capture safe)
    {
        long long n;
        n = (long long)N * F_IN;
        cvt_f32_to_bf16_kernel<<<(unsigned)((n / 4 + 255) / 256), blk, 0, stream>>>(x, x_bf, n);
        n = (long long)F_IN * HID;
        cvt_f32_to_bf16_kernel<<<(unsigned)((n / 4 + 255) / 256), blk, 0, stream>>>(emb_W, wb_emb, n);
        n = (long long)LAYERS * HID * HEADS * HID;
        cvt_f32_to_bf16_kernel<<<(unsigned)((n / 4 + 255) / 256), blk, 0, stream>>>(Wl, wb_Wl, n);
        cvt_f32_to_bf16_kernel<<<(unsigned)((n / 4 + 255) / 256), blk, 0, stream>>>(Wr, wb_Wr, n);
        n = (long long)HID * HID;
        cvt_f32_to_bf16_kernel<<<(unsigned)((n / 4 + 255) / 256), blk, 0, stream>>>(p1_W, wb_p1, n);
        n = (long long)HID * (HID / 2);
        cvt_f32_to_bf16_kernel<<<(unsigned)((n / 4 + 255) / 256), blk, 0, stream>>>(p2_W, wb_p2, n);
    }

    // embedding: h = relu(x @ emb_W + emb_b), plus bf16 mirror
    gemm_bf16_wmma<<<dim3(mblocks, HID / NT), blk, 0, stream>>>(
        x_bf, wb_emb, emb_b, h, h_bf, N, F_IN, HID, 1);

    const int ewave_blocks = (E * 32 + 255) / 256;
    const int eh_blocks    = (E * HEADS + 255) / 256;
    const int nh_blocks    = (N * HEADS + 255) / 256;
    const int nc_blocks    = (N * HID + 255) / 256;

    for (int L = 0; L < LAYERS; ++L) {
        fill_f32_kernel<<<nh_blocks, blk, 0, stream>>>(emax, (long long)N * HEADS, -INFINITY);
        fill_f32_kernel<<<nh_blocks, blk, 0, stream>>>(den,  (long long)N * HEADS, 0.0f);
        fill_f32_kernel<<<nc_blocks, blk, 0, stream>>>(acc,  (long long)N * HID,   0.0f);

        const unsigned short* WlL = wb_Wl + (size_t)L * HID * HEADS * HID;
        const unsigned short* WrL = wb_Wr + (size_t)L * HID * HEADS * HID;
        const float* blL = bl + (size_t)L * HEADS * HID;
        const float* brL = br + (size_t)L * HEADS * HID;
        const float* atL = att + (size_t)L * HEADS * HID;
        const float* cbL = cbias + (size_t)L * HID;

        gemm_bf16_wmma<<<dim3(mblocks, (HEADS * HID) / NT), blk, 0, stream>>>(
            h_bf, WlL, blL, xl, (unsigned short*)nullptr, N, HID, HEADS * HID, 0);
        gemm_bf16_wmma<<<dim3(mblocks, (HEADS * HID) / NT), blk, 0, stream>>>(
            h_bf, WrL, brL, xr, (unsigned short*)nullptr, N, HID, HEADS * HID, 0);

        edge_scores_kernel<<<ewave_blocks, blk, 0, stream>>>(
            ei, xl, xr, atL, ebuf, emax, E);
        edge_expsum_kernel<<<eh_blocks, blk, 0, stream>>>(
            ei, ebuf, emax, den, E);
        edge_aggregate_kernel<<<ewave_blocks, blk, 0, stream>>>(
            ei, xl, ebuf, den, acc, E);
        node_update_kernel<<<nc_blocks, blk, 0, stream>>>(h, h_bf, acc, cbL, N);
    }

    // predictor: z1 = relu(h @ p1_W + b)  (f32 into acc, bf16 mirror for p2)
    gemm_bf16_wmma<<<dim3(mblocks, HID / NT), blk, 0, stream>>>(
        h_bf, wb_p1, p1_b, acc, z1_bf, N, HID, HID, 1);
    // z2 = relu(z1 @ p2_W + b)            (f32 into ebuf: N*64 == E*4)
    gemm_bf16_wmma<<<dim3(mblocks, 1), blk, 0, stream>>>(
        z1_bf, wb_p2, p2_b, ebuf, (unsigned short*)nullptr, N, HID, HID / 2, 1);
    // scores
    p3_clip_kernel<<<(N + 255) / 256, blk, 0, stream>>>(
        ebuf, p3_W, p3_b, (float*)d_out, N);
}
